// MSRecurrentDecoder_6640019440442
// MI455X (gfx1250) — matrix operand order repaired and checked by reading, not compiled, not run on hardware
//
#include <hip/hip_runtime.h>
#include <hip/hip_bf16.h>

typedef _Float16 half_t;
typedef __attribute__((ext_vector_type(16))) _Float16 v16h;
typedef __attribute__((ext_vector_type(8)))  _Float16 v8h;
typedef __attribute__((ext_vector_type(8)))  float    v8f;

// ---------------------------------------------------------------------------
// prep: X[b][y][x][c] (f16, channels-last, padded to CINP) =
//       concat(feat0[C], flow[2], warp(feat1, flow)[C]), zero pad to CINP.
// ---------------------------------------------------------------------------
template<int C, int CINP>
__global__ void prep_kernel(const float* __restrict__ feat0,
                            const float* __restrict__ feat1,
                            const float* __restrict__ flow,
                            half_t* __restrict__ X,
                            int Bn, int H, int W) {
  int idx = blockIdx.x * blockDim.x + threadIdx.x;
  int total = Bn * H * W;
  if (idx >= total) return;
  int x = idx % W;
  int y = (idx / W) % H;
  int b = idx / (W * H);
  const size_t HW = (size_t)H * W;
  half_t* xp = X + (size_t)idx * CINP;
  const float* f0 = feat0 + (size_t)b * C * HW + (size_t)y * W + x;
  float fx = flow[((size_t)b * 2 + 0) * HW + (size_t)y * W + x];
  float fy = flow[((size_t)b * 2 + 1) * HW + (size_t)y * W + x];
  for (int c = 0; c < C; ++c) xp[c] = (half_t)f0[c * HW];
  xp[C]     = (half_t)fx;
  xp[C + 1] = (half_t)fy;
  // backward warp, bilinear, zeros outside
  float gx = (float)x + fx, gy = (float)y + fy;
  float x0f = floorf(gx), y0f = floorf(gy);
  float wx = gx - x0f, wy = gy - y0f;
  int xi = (int)x0f, yi = (int)y0f;
  bool vx0 = (xi >= 0) && (xi <= W - 1);
  bool vx1 = (xi + 1 >= 0) && (xi + 1 <= W - 1);
  bool vy0 = (yi >= 0) && (yi <= H - 1);
  bool vy1 = (yi + 1 >= 0) && (yi + 1 <= H - 1);
  int xc0 = min(max(xi, 0), W - 1);
  int xc1 = min(max(xi + 1, 0), W - 1);
  int yc0 = min(max(yi, 0), H - 1);
  int yc1 = min(max(yi + 1, 0), H - 1);
  float w00 = (1.f - wx) * (1.f - wy) * ((vx0 && vy0) ? 1.f : 0.f);
  float w01 = wx * (1.f - wy) * ((vx1 && vy0) ? 1.f : 0.f);
  float w10 = (1.f - wx) * wy * ((vx0 && vy1) ? 1.f : 0.f);
  float w11 = wx * wy * ((vx1 && vy1) ? 1.f : 0.f);
  size_t i00 = (size_t)yc0 * W + xc0, i01 = (size_t)yc0 * W + xc1;
  size_t i10 = (size_t)yc1 * W + xc0, i11 = (size_t)yc1 * W + xc1;
  const float* f1 = feat1 + (size_t)b * C * HW;
  for (int c = 0; c < C; ++c) {
    const float* p = f1 + (size_t)c * HW;
    float v = p[i00] * w00 + p[i01] * w01 + p[i10] * w10 + p[i11] * w11;
    xp[C + 2 + c] = (half_t)v;
  }
  for (int c = 2 * C + 2; c < CINP; ++c) xp[c] = (half_t)0.f;
}

// ---------------------------------------------------------------------------
// Weight conversion: [O][CIN][3][3] f32 -> [O][9*CINP] f16, k = tap*CINP + c
// ---------------------------------------------------------------------------
template<int OUTC, int CIN, int CINP>
__global__ void cvt_wa_kernel(const float* __restrict__ src, half_t* __restrict__ dst) {
  constexpr int KTOT = 9 * CINP;
  int idx = blockIdx.x * blockDim.x + threadIdx.x;
  if (idx >= OUTC * KTOT) return;
  int o = idx / KTOT;
  int k = idx % KTOT;
  int t = k / CINP;
  int c = k % CINP;
  half_t v = (half_t)0.f;
  if (c < CIN) v = (half_t)src[((size_t)o * CIN + c) * 9 + t];
  dst[idx] = v;
}

// [2][CIN][3][3] f32 -> [16][9*CIN] f16 (rows 2..15 zero), k = tap*CIN + c
template<int CIN>
__global__ void cvt_wb_kernel(const float* __restrict__ src, half_t* __restrict__ dst) {
  constexpr int KTOT = 9 * CIN;
  int idx = blockIdx.x * blockDim.x + threadIdx.x;
  if (idx >= 16 * KTOT) return;
  int o = idx / KTOT;
  int k = idx % KTOT;
  int t = k / CIN;
  int c = k % CIN;
  half_t v = (half_t)0.f;
  if (o < 2) v = (half_t)src[((size_t)o * CIN + c) * 9 + t];
  dst[idx] = v;
}

// ---------------------------------------------------------------------------
// convA: h = GELU(conv3x3(X) + bias), implicit GEMM via WMMA f16->f32.
// Block = (COUT/16) waves; wave w computes out-channels [16w,16w+16) x 16 px.
// LDS holds a 3 x 18 x CINP f16 input patch (zero halo).
// ---------------------------------------------------------------------------
template<int COUT, int CINP>
__global__ void __launch_bounds__((COUT / 16) * 32)
conv_a_kernel(const half_t* __restrict__ X, const half_t* __restrict__ Wt,
              const float* __restrict__ bias, half_t* __restrict__ Hh,
              int Bn, int H, int W) {
  constexpr int KTOT = 9 * CINP;
  constexpr int LDSH = 3 * 18 * CINP;
  __shared__ __attribute__((aligned(64))) half_t lds[LDSH];

  int tilesX = W / 16;
  int tile = blockIdx.x;
  int x0 = (tile % tilesX) * 16;
  int y  = (tile / tilesX) % H;
  int b  = tile / (tilesX * H);

  // Cooperative stage of the input patch (rows y-1..y+1, cols x0-1..x0+16).
  int tid = threadIdx.x;
  int nthr = blockDim.x;
  for (int v = tid; v < LDSH / 8; v += nthr) {
    int h = v * 8;
    int c = h % CINP;
    int col = (h / CINP) % 18;
    int row = h / (CINP * 18);
    int yy = y + row - 1;
    int xx = x0 + col - 1;
    v8h val = {};
    if (yy >= 0 && yy < H && xx >= 0 && xx < W) {
      val = *(const v8h*)(X + (((size_t)(b * H + yy) * W + xx) * CINP + c));
    }
    *(v8h*)(lds + h) = val;
  }
  __syncthreads();

  int lane = tid & 31;
  int wave = tid >> 5;
  int n = lane & 15;   // pixel column of the tile (A-row index uses same value)
  int g = lane >> 4;   // half-wave group
  int oc = wave * 16 + n;

  v8f acc = {};
  const half_t* wrow = Wt + (size_t)oc * KTOT;
  for (int t = 0; t < 9; ++t) {
    int ky = t / 3, kx = t % 3;
    const half_t* abase = wrow + t * CINP + g * 16;
    const half_t* bbase = lds + (ky * 18 + n + kx) * CINP + g * 16;
#pragma unroll
    for (int c0 = 0; c0 < CINP; c0 += 32) {
      v16h a = *(const v16h*)(abase + c0);       // K = c0 + g*16 + [0..15]
      v16h bf = *(const v16h*)(bbase + c0);      // K = c0 + g*16 + [0..15]
      acc = __builtin_amdgcn_wmma_f32_16x16x32_f16(
          false, a, false, bf, (short)0, acc, false, false);
    }
  }

  // Epilogue: bias + exact GELU, store 8 consecutive channels per lane.
  int x = x0 + n;
  v8h out;
#pragma unroll
  for (int r = 0; r < 8; ++r) {
    int ch = wave * 16 + 8 * g + r;        // D row M = r + 8*g
    float v = acc[r] + bias[ch];
    v = 0.5f * v * (1.f + erff(v * 0.70710678118654752f));
    out[r] = (half_t)v;
  }
  *(v8h*)(Hh + (((size_t)(b * H + y) * W + x) * COUT + wave * 16 + 8 * g)) = out;
}

// ---------------------------------------------------------------------------
// convB: flow_out = flow_in + conv3x3(h) + bias  (2 out channels, M padded 16)
// One wave per 16-pixel tile.
// ---------------------------------------------------------------------------
template<int CIN>
__global__ void __launch_bounds__(32)
conv_b_kernel(const half_t* __restrict__ Hh, const half_t* __restrict__ Wt,
              const float* __restrict__ bias, const float* __restrict__ flow_in,
              float* __restrict__ flow_out, int Bn, int H, int W) {
  constexpr int KTOT = 9 * CIN;
  constexpr int LDSH = 3 * 18 * CIN;
  __shared__ __attribute__((aligned(64))) half_t lds[LDSH];

  int tilesX = W / 16;
  int tile = blockIdx.x;
  int x0 = (tile % tilesX) * 16;
  int y  = (tile / tilesX) % H;
  int b  = tile / (tilesX * H);

  int tid = threadIdx.x;
  for (int v = tid; v < LDSH / 8; v += 32) {
    int h = v * 8;
    int c = h % CIN;
    int col = (h / CIN) % 18;
    int row = h / (CIN * 18);
    int yy = y + row - 1;
    int xx = x0 + col - 1;
    v8h val = {};
    if (yy >= 0 && yy < H && xx >= 0 && xx < W) {
      val = *(const v8h*)(Hh + (((size_t)(b * H + yy) * W + xx) * CIN + c));
    }
    *(v8h*)(lds + h) = val;
  }
  __syncthreads();

  int lane = tid & 31;
  int n = lane & 15;
  int g = lane >> 4;

  v8f acc = {};
  const half_t* wrow = Wt + (size_t)n * KTOT;  // A row = n (rows >=2 are zero)
  for (int t = 0; t < 9; ++t) {
    int ky = t / 3, kx = t % 3;
    const half_t* abase = wrow + t * CIN + g * 16;
    const half_t* bbase = lds + (ky * 18 + n + kx) * CIN + g * 16;
#pragma unroll
    for (int c0 = 0; c0 < CIN; c0 += 32) {
      v16h a = *(const v16h*)(abase + c0);
      v16h bf = *(const v16h*)(bbase + c0);
      acc = __builtin_amdgcn_wmma_f32_16x16x32_f16(
          false, a, false, bf, (short)0, acc, false, false);
    }
  }

  if (g == 0) {  // lanes 0..15 hold rows M=0 (acc[0]) and M=1 (acc[1])
    int x = x0 + n;
    size_t hw = (size_t)H * W;
    size_t base = (size_t)b * 2 * hw + (size_t)y * W + x;
    flow_out[base]      = flow_in[base]      + acc[0] + bias[0];
    flow_out[base + hw] = flow_in[base + hw] + acc[1] + bias[1];
  }
}

// ---------------------------------------------------------------------------
// 2x bilinear upsample, align_corners=True
// ---------------------------------------------------------------------------
__global__ void upsample_kernel(const float* __restrict__ in, float* __restrict__ out,
                                int Hin, int Win, int Hout, int Wout, int BC) {
  int idx = blockIdx.x * blockDim.x + threadIdx.x;
  int total = BC * Hout * Wout;
  if (idx >= total) return;
  int x = idx % Wout;
  int y = (idx / Wout) % Hout;
  int bc = idx / (Wout * Hout);
  float cy = (float)y * (float)(Hin - 1) / (float)(Hout - 1);
  float cx = (float)x * (float)(Win - 1) / (float)(Wout - 1);
  int i0 = min(max((int)floorf(cy), 0), Hin - 2);
  int j0 = min(max((int)floorf(cx), 0), Win - 2);
  float wy = cy - (float)i0;
  float wx = cx - (float)j0;
  const float* p = in + (size_t)bc * Hin * Win;
  float v = (p[(size_t)i0 * Win + j0] * (1.f - wx) + p[(size_t)i0 * Win + j0 + 1] * wx) * (1.f - wy)
          + (p[(size_t)(i0 + 1) * Win + j0] * (1.f - wx) + p[(size_t)(i0 + 1) * Win + j0 + 1] * wx) * wy;
  out[idx] = v;
}

// ---------------------------------------------------------------------------
// Orchestration
// ---------------------------------------------------------------------------
extern "C" void kernel_launch(void* const* d_in, const int* in_sizes, int n_in,
                              void* d_out, int out_size, void* d_ws, size_t ws_size,
                              hipStream_t stream) {
  const float* flow0  = (const float*)d_in[0];
  const float* lvl1_0 = (const float*)d_in[1];
  const float* lvl1_1 = (const float*)d_in[2];
  const float* lvl2_0 = (const float*)d_in[3];
  const float* lvl2_1 = (const float*)d_in[4];
  const float* w1a = (const float*)d_in[5];
  const float* b1a = (const float*)d_in[6];
  const float* w1b = (const float*)d_in[7];
  const float* b1b = (const float*)d_in[8];
  const float* w0a = (const float*)d_in[9];
  const float* b0a = (const float*)d_in[10];
  const float* w0b = (const float*)d_in[11];
  const float* b0b = (const float*)d_in[12];

  constexpr int B = 4;
  constexpr int C1 = 128, H8 = 96,  W8 = 128;   // coarse
  constexpr int C0 = 64,  H4 = 192, W4 = 256;   // fine
  constexpr int CINP2 = 288;   // pad(2*128+2=258) to mult of 32
  constexpr int CINP1 = 160;   // pad(2*64+2=130) to mult of 32

  char* ws = (char*)d_ws;
  size_t off = 0;
  auto take = [&](size_t bytes) -> char* {
    char* p = ws + off;
    off = (off + bytes + 255) & ~(size_t)255;
    return p;
  };
  half_t* Xbuf = (half_t*)take((size_t)B * H4 * W4 * CINP1 * 2);   // also covers lvl2 X
  half_t* Hbuf = (half_t*)take((size_t)B * H4 * W4 * C0 * 2);      // also covers lvl2 h
  half_t* wA2  = (half_t*)take((size_t)C1 * 9 * CINP2 * 2);
  half_t* wB2  = (half_t*)take((size_t)16 * 9 * C1 * 2);
  half_t* wA1  = (half_t*)take((size_t)C0 * 9 * CINP1 * 2);
  half_t* wB1  = (half_t*)take((size_t)16 * 9 * C0 * 2);
  float*  f2A  = (float*)take((size_t)B * 2 * H8 * W8 * 4);
  float*  f2B  = (float*)take((size_t)B * 2 * H8 * W8 * 4);
  float*  f1A  = (float*)take((size_t)B * 2 * H4 * W4 * 4);
  float*  f1B  = (float*)take((size_t)B * 2 * H4 * W4 * 4);
  if (off > ws_size) return;  // workspace too small; nothing safe to do

  // --- weight conversion (f32 -> WMMA-layout f16) ---
  {
    int n2 = C1 * 9 * CINP2;
    cvt_wa_kernel<C1, 2 * C1 + 2, CINP2><<<(n2 + 255) / 256, 256, 0, stream>>>(w1a, wA2);
    int nb2 = 16 * 9 * C1;
    cvt_wb_kernel<C1><<<(nb2 + 255) / 256, 256, 0, stream>>>(w1b, wB2);
    int n1 = C0 * 9 * CINP1;
    cvt_wa_kernel<C0, 2 * C0 + 2, CINP1><<<(n1 + 255) / 256, 256, 0, stream>>>(w0a, wA1);
    int nb1 = 16 * 9 * C0;
    cvt_wb_kernel<C0><<<(nb1 + 255) / 256, 256, 0, stream>>>(w0b, wB1);
  }

  // --- level 2 (coarse): 4 refinement iterations ---
  {
    int npix = B * H8 * W8;
    int ntiles = B * H8 * (W8 / 16);
    const float* fin = flow0;
    for (int it = 0; it < 4; ++it) {
      float* fout = (it & 1) ? f2B : f2A;
      if (it == 3) fout = f2B;           // keep final result in f2B
      else if (it == 2) fout = f2A;
      prep_kernel<C1, CINP2><<<(npix + 255) / 256, 256, 0, stream>>>(
          lvl2_0, lvl2_1, fin, Xbuf, B, H8, W8);
      conv_a_kernel<C1, CINP2><<<ntiles, (C1 / 16) * 32, 0, stream>>>(
          Xbuf, wA2, b1a, Hbuf, B, H8, W8);
      conv_b_kernel<C1><<<ntiles, 32, 0, stream>>>(
          Hbuf, wB2, b1b, fin, fout, B, H8, W8);
      fin = fout;
    }
    // upsample coarse flow to fine resolution into f1A
    int nout = B * 2 * H4 * W4;
    upsample_kernel<<<(nout + 255) / 256, 256, 0, stream>>>(
        fin, f1A, H8, W8, H4, W4, B * 2);
  }

  // --- level 1 (fine): 4 refinement iterations, last writes d_out ---
  {
    int npix = B * H4 * W4;
    int ntiles = B * H4 * (W4 / 16);
    const float* fin = f1A;
    for (int it = 0; it < 4; ++it) {
      float* fout = (it == 3) ? (float*)d_out : ((it & 1) ? f1A : f1B);
      prep_kernel<C0, CINP1><<<(npix + 255) / 256, 256, 0, stream>>>(
          lvl1_0, lvl1_1, fin, Xbuf, B, H4, W4);
      conv_a_kernel<C0, CINP1><<<ntiles, (C0 / 16) * 32, 0, stream>>>(
          Xbuf, wA1, b0a, Hbuf, B, H4, W4);
      conv_b_kernel<C0><<<ntiles, 32, 0, stream>>>(
          Hbuf, wB1, b0b, fin, fout, B, H4, W4);
      fin = fout;
    }
  }
}
